// MRFMessagePassingModule_82343112999097
// MI455X (gfx1250) — matrix operand order repaired and checked by reading, not compiled, not run on hardware
//
#include <hip/hip_runtime.h>
#include <hip/hip_bf16.h>

#define EDGES   16896
#define NTOT    1536
#define NGRAPH  128
#define EPG     132
#define NPG     12
#define NACT    16
#define MIDP    32
#define KE      192
#define KN      128
#define NOUT    512

typedef __attribute__((ext_vector_type(16))) _Float16 v16h;
typedef __attribute__((ext_vector_type(8)))  _Float16 v8h;
typedef __attribute__((ext_vector_type(8)))  float    v8f;

// ---------------------------------------------------------------------------
// K1: build f16 concatenated edge activations xe = [edge_feats | u_e]
// ---------------------------------------------------------------------------
__global__ void k_prep_xe(const float* __restrict__ ef, const float* __restrict__ efr,
                          const float* __restrict__ gf,
                          _Float16* __restrict__ xe, _Float16* __restrict__ xer) {
    int t = blockIdx.x * blockDim.x + threadIdx.x;
    if (t >= EDGES * KE) return;
    int e = t / KE, k = t % KE;
    int g = e / EPG;
    float ve = (k < 128) ? ef[e * 128 + k] : gf[g * 64 + (k - 128)];
    float vr = (k < 128) ? efr[e * 128 + k] : gf[g * 64 + (k - 128)];
    xe[t]  = (_Float16)ve;
    xer[t] = (_Float16)vr;
}

// ---------------------------------------------------------------------------
// K2: pack W (192x512, row-major K-major) into WMMA B-fragment layout.
// Fragment f = kstep*32 + ntile; lane<16: K=kbase+h, N=ntile*16+lane
//                                lane>=16: K=kbase+16+h, N=ntile*16+lane-16
// ---------------------------------------------------------------------------
__global__ void k_pack_w(const float* __restrict__ W, _Float16* __restrict__ Wp) {
    int t = blockIdx.x * blockDim.x + threadIdx.x;
    if (t >= 6 * 32 * 512) return;
    int frag = t / 512;
    int lane = (t % 512) / 16;
    int h    = t % 16;
    int kstep = frag / 32, ntile = frag % 32;
    int k = kstep * 32 + ((lane < 16) ? h : 16 + h);
    int n = ntile * 16 + (lane & 15);
    Wp[t] = (_Float16)W[k * NOUT + n];
}

// ---------------------------------------------------------------------------
// K3: edge GEMM  out = xe @ W + bias   (and reflected variant sharing B-frags)
// block = 128 (4 waves), one M-tile (16 edges) per block, wave w covers
// ntiles w*8..w*8+7.  6 K-steps of v_wmma_f32_16x16x32_f16.
// ---------------------------------------------------------------------------
__global__ void k_gemm_edge(const _Float16* __restrict__ xe, const _Float16* __restrict__ xer,
                            const _Float16* __restrict__ Wp, const float* __restrict__ bias,
                            _Float16* __restrict__ outF, _Float16* __restrict__ outR) {
    int mtile = blockIdx.x;
    int wave  = threadIdx.x >> 5;
    int lane  = threadIdx.x & 31;
    int row   = lane & 15;
    int hi    = lane >> 4;

    v16h A[6], Ar[6];
    {
        const _Float16* pa  = xe  + (size_t)(mtile * 16 + row) * KE;
        const _Float16* par = xer + (size_t)(mtile * 16 + row) * KE;
#pragma unroll
        for (int ks = 0; ks < 6; ++ks) {
            int b0 = ks * 32 + (hi ? 16 : 0);
            int b1 = ks * 32 + (hi ? 24 : 8);
            v8h x0 = *(const v8h*)(pa + b0);
            v8h x1 = *(const v8h*)(pa + b1);
            v8h y0 = *(const v8h*)(par + b0);
            v8h y1 = *(const v8h*)(par + b1);
#pragma unroll
            for (int i = 0; i < 8; ++i) {
                A[ks][i] = x0[i];  A[ks][8 + i] = x1[i];
                Ar[ks][i] = y0[i]; Ar[ks][8 + i] = y1[i];
            }
        }
    }

    for (int nt0 = 0; nt0 < 8; ++nt0) {
        int ntile = wave * 8 + nt0;
        float bv = bias[ntile * 16 + row];
        v8f accF, accR;
#pragma unroll
        for (int i = 0; i < 8; ++i) { accF[i] = bv; accR[i] = bv; }
#pragma unroll
        for (int ks = 0; ks < 6; ++ks) {
            const _Float16* pb = Wp + ((size_t)(ks * 32 + ntile) * 512) + lane * 16;
            v16h b = *(const v16h*)pb;
            accF = __builtin_amdgcn_wmma_f32_16x16x32_f16(false, A[ks],  false, b,
                                                          (short)0, accF, false, false);
            accR = __builtin_amdgcn_wmma_f32_16x16x32_f16(false, Ar[ks], false, b,
                                                          (short)0, accR, false, false);
        }
#pragma unroll
        for (int v = 0; v < 8; ++v) {
            int er = mtile * 16 + v + hi * 8;
            int n  = ntile * 16 + row;
            outF[(size_t)er * NOUT + n] = (_Float16)accF[v];
            outR[(size_t)er * NOUT + n] = (_Float16)accR[v];
        }
    }
}

// ---------------------------------------------------------------------------
// K4: per-edge einsum  vals[e,a,b] = sum_m fc[e,a,m] * fcr[e,b,m]
// One wave per edge: A = fc_e (16x32), B[k=m][n=b] = fcr_e[b][m]; 1 WMMA.
// ---------------------------------------------------------------------------
__global__ void k_einsum(const _Float16* __restrict__ fc, const _Float16* __restrict__ fcr,
                         float* __restrict__ vals) {
    int wave = threadIdx.x >> 5;
    int lane = threadIdx.x & 31;
    int e = blockIdx.x * (blockDim.x >> 5) + wave;
    if (e >= EDGES) return;
    int row = lane & 15, hi = lane >> 4;

    const _Float16* pa = fc  + (size_t)e * 512 + row * 32;
    const _Float16* pb = fcr + (size_t)e * 512 + row * 32;
    v8h a0 = *(const v8h*)(pa + (hi ? 8 : 0));
    v8h a1 = *(const v8h*)(pa + (hi ? 24 : 16));
    v8h b0 = *(const v8h*)(pb + (hi ? 8 : 0));
    v8h b1 = *(const v8h*)(pb + (hi ? 24 : 16));
    v16h a, b;
#pragma unroll
    for (int i = 0; i < 8; ++i) {
        a[i] = a0[i]; a[8 + i] = a1[i];
        b[i] = b0[i]; b[8 + i] = b1[i];
    }
    v8f acc = {};
    acc = __builtin_amdgcn_wmma_f32_16x16x32_f16(false, a, false, b,
                                                 (short)0, acc, false, false);
#pragma unroll
    for (int v = 0; v < 8; ++v) {
        int arow = v + hi * 8;
        vals[(size_t)e * 256 + arow * 16 + row] = acc[v];
    }
}

// ---------------------------------------------------------------------------
// K5: node projections indiv_factor / indiv_util  [Nt,16] (tiny, scalar f32)
// ---------------------------------------------------------------------------
__global__ void k_node_proj(const float* __restrict__ nf, const float* __restrict__ gf,
                            const float* __restrict__ Wif, const float* __restrict__ bif,
                            const float* __restrict__ Wiu, const float* __restrict__ biu,
                            float* __restrict__ indf, float* __restrict__ indu) {
    int t = blockIdx.x * blockDim.x + threadIdx.x;
    if (t >= NTOT * NACT) return;
    int n = t / NACT, a = t % NACT;
    int g = n / NPG;
    float sf = bif[a], su = biu[a];
    for (int k = 0; k < KN; ++k) {
        float x = (k < 64) ? nf[n * 64 + k] : gf[g * 64 + (k - 64)];
        sf += x * Wif[k * NACT + a];
        su += x * Wiu[k * NACT + a];
    }
    indf[t] = sf;
    indu[t] = su;
}

// ---------------------------------------------------------------------------
// K6: straight-through gumbel gate -> forward value is y_hard:
// flag0[e] = 1 if (logit0+noise0 >= logit1+noise1) else 0
// ---------------------------------------------------------------------------
__global__ void k_flag(const float* __restrict__ ef, const float* __restrict__ efr,
                       const float* __restrict__ gf,
                       const float* __restrict__ Wfl, const float* __restrict__ bfl,
                       const float* __restrict__ Wcls, const float* __restrict__ bcls,
                       const int* __restrict__ idx_revs, const float* __restrict__ gnoise,
                       float* __restrict__ flag0) {
    int e = blockIdx.x * blockDim.x + threadIdx.x;
    if (e >= EDGES) return;
    int g = e / EPG;
    float l0 = bcls[0], l1 = bcls[1];
    for (int j = 0; j < MIDP; ++j) {
        float f = bfl[j], fr = bfl[j];
        for (int k = 0; k < KE; ++k) {
            float w  = Wfl[k * MIDP + j];
            float x  = (k < 128) ? ef[e * 128 + k]  : gf[g * 64 + (k - 128)];
            float xr = (k < 128) ? efr[e * 128 + k] : gf[g * 64 + (k - 128)];
            f  += x * w;
            fr += xr * w;
        }
        float rep = f * fr;
        l0 += rep * Wcls[j * 2 + 0];
        l1 += rep * Wcls[j * 2 + 1];
    }
    int r  = idx_revs[e];
    int ng = (e < r) ? e : r;
    float n0 = gnoise[ng * 2 + 0], n1 = gnoise[ng * 2 + 1];
    flag0[e] = ((l0 + n0) >= (l1 + n1)) ? 1.0f : 0.0f;
}

// ---------------------------------------------------------------------------
// K7: zero-init
// ---------------------------------------------------------------------------
__global__ void k_zero(float* __restrict__ p, int n) {
    int t = blockIdx.x * blockDim.x + threadIdx.x;
    if (t < n) p[t] = 0.0f;
}

// ---------------------------------------------------------------------------
// K8: one BP step over edges, thread = (edge, direction)
// dir 0 (fwd):  m = F + agg[b],  F row-major; writes nmsg/numsg
// dir 1 (rev):  m = F^T + agg[b];             writes nrmsg/nrumsg
// ---------------------------------------------------------------------------
__global__ void k_bp_edge(const float* __restrict__ fv, const float* __restrict__ uv,
                          const float* __restrict__ mm, const float* __restrict__ um,
                          const float* __restrict__ msg, const float* __restrict__ rmsg,
                          const float* __restrict__ umsg, const float* __restrict__ rumsg,
                          const float* __restrict__ indf, const float* __restrict__ indu,
                          const float* __restrict__ flag0,
                          const int* __restrict__ src, const int* __restrict__ dst,
                          float* __restrict__ nmsg, float* __restrict__ nrmsg,
                          float* __restrict__ numsg, float* __restrict__ nrumsg) {
    int t = blockIdx.x * blockDim.x + threadIdx.x;
    if (t >= 2 * EDGES) return;
    int dir = t >= EDGES;
    int e   = t - dir * EDGES;

    int node = dir ? dst[e] : src[e];
    const float* sub  = dir ? msg  : rmsg;
    const float* usub = dir ? umsg : rumsg;

    float agg[16], util[16];
#pragma unroll
    for (int b = 0; b < 16; ++b) {
        agg[b]  = mm[node * 16 + b] - sub[(size_t)e * 16 + b]  + indf[node * 16 + b];
        util[b] = um[node * 16 + b] - usub[(size_t)e * 16 + b] + indu[node * 16 + b];
    }

    const float* F = fv + (size_t)e * 256;
    const float* U = uv + (size_t)e * 256;

    float s[16], uo[16];
    float cmax = -3.0e38f;
#pragma unroll
    for (int a = 0; a < 16; ++a) {
        float mx = -3.0e38f;
#pragma unroll
        for (int b = 0; b < 16; ++b) {
            float f = dir ? F[b * 16 + a] : F[a * 16 + b];
            float v = f + agg[b];
            mx = fmaxf(mx, v);
        }
        float se = 0.0f, num = 0.0f;
#pragma unroll
        for (int b = 0; b < 16; ++b) {
            float f  = dir ? F[b * 16 + a] : F[a * 16 + b];
            float uu = dir ? U[b * 16 + a] : U[a * 16 + b];
            float p  = __expf(f + agg[b] - mx);
            se  += p;
            num += (uu + util[b]) * p;
        }
        s[a]  = mx + __logf(se);
        uo[a] = num / se;
        cmax = fmaxf(cmax, s[a]);
    }
    float cs = 0.0f;
#pragma unroll
    for (int a = 0; a < 16; ++a) cs += __expf(s[a] - cmax);
    float clse = cmax + __logf(cs);

    float fl = flag0[e];
    float* om = dir ? nrmsg  : nmsg;
    float* ou = dir ? nrumsg : numsg;
#pragma unroll
    for (int a = 0; a < 16; ++a) {
        om[(size_t)e * 16 + a] = (s[a] - clse) * fl;
        ou[(size_t)e * 16 + a] = uo[a] * fl;
    }
}

// ---------------------------------------------------------------------------
// K9: gather-style segment_sum over dst using closed-form FC-graph indexing:
// edge (s -> d) in graph g has id  g*132 + s*11 + (d - (d>s))
// ---------------------------------------------------------------------------
__global__ void k_aggregate(const float* __restrict__ nmsg, const float* __restrict__ numsg,
                            float* __restrict__ mm, float* __restrict__ um) {
    int t = blockIdx.x * blockDim.x + threadIdx.x;
    if (t >= NTOT * NACT) return;
    int n = t / NACT, b = t % NACT;
    int g = n / NPG, ln = n % NPG;
    float sm = 0.0f, su = 0.0f;
#pragma unroll
    for (int sN = 0; sN < NPG; ++sN) {
        if (sN == ln) continue;
        int d   = ln - (ln > sN ? 1 : 0);
        int eid = g * EPG + sN * (NPG - 1) + d;
        sm += nmsg[(size_t)eid * 16 + b];
        su += numsg[(size_t)eid * 16 + b];
    }
    mm[t] = sm;
    um[t] = su;
}

// ---------------------------------------------------------------------------
// K10: output  (um_n + indiv_util)[node0]  -> [128,16]
// ---------------------------------------------------------------------------
__global__ void k_output(const float* __restrict__ um, const float* __restrict__ indu,
                         float* __restrict__ out) {
    int t = blockIdx.x * blockDim.x + threadIdx.x;
    if (t >= NGRAPH * NACT) return;
    int g = t / NACT, a = t % NACT;
    int n = g * NPG;
    out[t] = um[n * 16 + a] + indu[n * 16 + a];
}

// ---------------------------------------------------------------------------
// workspace layout (bytes, all offsets 256B aligned)
// ---------------------------------------------------------------------------
static const size_t SZ_XE    = (size_t)EDGES * KE * 2;       // 6,488,064
static const size_t SZ_WP    = (size_t)6 * 32 * 512 * 2;     // 196,608
static const size_t SZ_NA    = (size_t)NTOT * NACT * 4;      // 98,304
static const size_t SZ_FLAG  = (size_t)EDGES * 4;            // 67,584
static const size_t SZ_VAL   = (size_t)EDGES * 256 * 4;      // 17,301,504
static const size_t SZ_FC    = (size_t)EDGES * NOUT * 2;     // 17,301,504
static const size_t SZ_EA    = (size_t)EDGES * NACT * 4;     // 1,081,344

static const size_t OFF_XE    = 0;
static const size_t OFF_XER   = OFF_XE + SZ_XE;
static const size_t OFF_WPJF  = OFF_XER + SZ_XE;
static const size_t OFF_WPJU  = OFF_WPJF + SZ_WP;
static const size_t OFF_IF    = OFF_WPJU + SZ_WP;
static const size_t OFF_IU    = OFF_IF + SZ_NA;
static const size_t OFF_FLAG  = OFF_IU + SZ_NA;
static const size_t OFF_FV    = OFF_FLAG + SZ_FLAG;
static const size_t OFF_UV    = OFF_FV + SZ_VAL;
static const size_t OFF_R     = OFF_UV + SZ_VAL;   // reusable region
static const size_t OFF_FC    = OFF_R;
static const size_t OFF_FCR   = OFF_FC + SZ_FC;
// BP state aliases the (dead after einsums) FC region:
static const size_t OFF_MM    = OFF_R;
static const size_t OFF_UM    = OFF_MM + 2 * SZ_NA;
static const size_t OFF_MSG   = OFF_UM + 2 * SZ_NA;
static const size_t OFF_RMSG  = OFF_MSG + 2 * SZ_EA;
static const size_t OFF_UMSG  = OFF_RMSG + 2 * SZ_EA;
static const size_t OFF_RUMSG = OFF_UMSG + 2 * SZ_EA;
static const size_t BP_FLOATS = (4 * SZ_NA + 8 * SZ_EA) / 4;

extern "C" void kernel_launch(void* const* d_in, const int* in_sizes, int n_in,
                              void* d_out, int out_size, void* d_ws, size_t ws_size,
                              hipStream_t stream) {
    const float* edge_feats  = (const float*)d_in[0];
    const float* node_feats  = (const float*)d_in[1];
    const float* graph_feats = (const float*)d_in[2];
    const float* edge_refl   = (const float*)d_in[3];
    const int*   src         = (const int*)d_in[4];
    const int*   dst         = (const int*)d_in[5];
    const int*   idx_revs    = (const int*)d_in[6];
    const float* gnoise      = (const float*)d_in[7];
    const float* W_jf = (const float*)d_in[8];
    const float* b_jf = (const float*)d_in[9];
    const float* W_if = (const float*)d_in[10];
    const float* b_if = (const float*)d_in[11];
    const float* W_ju = (const float*)d_in[12];
    const float* b_ju = (const float*)d_in[13];
    const float* W_iu = (const float*)d_in[14];
    const float* b_iu = (const float*)d_in[15];
    const float* W_fl = (const float*)d_in[16];
    const float* b_fl = (const float*)d_in[17];
    const float* W_cls = (const float*)d_in[18];
    const float* b_cls = (const float*)d_in[19];

    char* ws = (char*)d_ws;
    _Float16* xe   = (_Float16*)(ws + OFF_XE);
    _Float16* xer  = (_Float16*)(ws + OFF_XER);
    _Float16* WpJF = (_Float16*)(ws + OFF_WPJF);
    _Float16* WpJU = (_Float16*)(ws + OFF_WPJU);
    float* indf  = (float*)(ws + OFF_IF);
    float* indu  = (float*)(ws + OFF_IU);
    float* flag0 = (float*)(ws + OFF_FLAG);
    float* fv    = (float*)(ws + OFF_FV);
    float* uvv   = (float*)(ws + OFF_UV);
    _Float16* fc  = (_Float16*)(ws + OFF_FC);
    _Float16* fcr = (_Float16*)(ws + OFF_FCR);
    float* mmB[2]   = { (float*)(ws + OFF_MM),            (float*)(ws + OFF_MM + SZ_NA) };
    float* umB[2]   = { (float*)(ws + OFF_UM),            (float*)(ws + OFF_UM + SZ_NA) };
    float* msgB[2]  = { (float*)(ws + OFF_MSG),           (float*)(ws + OFF_MSG + SZ_EA) };
    float* rmsgB[2] = { (float*)(ws + OFF_RMSG),          (float*)(ws + OFF_RMSG + SZ_EA) };
    float* umsgB[2] = { (float*)(ws + OFF_UMSG),          (float*)(ws + OFF_UMSG + SZ_EA) };
    float* rumsgB[2]= { (float*)(ws + OFF_RUMSG),         (float*)(ws + OFF_RUMSG + SZ_EA) };

    // ---- feature prep + weight packing ----
    k_prep_xe<<<(EDGES * KE) / 256, 256, 0, stream>>>(edge_feats, edge_refl, graph_feats, xe, xer);
    k_pack_w<<<(6 * 32 * 512) / 256, 256, 0, stream>>>(W_jf, WpJF);
    k_pack_w<<<(6 * 32 * 512) / 256, 256, 0, stream>>>(W_ju, WpJU);

    // ---- factor path: GEMM (WMMA) then per-edge einsum (WMMA) ----
    k_gemm_edge<<<EDGES / 16, 128, 0, stream>>>(xe, xer, WpJF, b_jf, fc, fcr);
    k_einsum<<<EDGES / 4, 128, 0, stream>>>(fc, fcr, fv);

    // ---- util path: reuse fc/fcr staging buffers ----
    k_gemm_edge<<<EDGES / 16, 128, 0, stream>>>(xe, xer, WpJU, b_ju, fc, fcr);
    k_einsum<<<EDGES / 4, 128, 0, stream>>>(fc, fcr, uvv);

    // ---- small projections + gate ----
    k_node_proj<<<(NTOT * NACT) / 256, 256, 0, stream>>>(node_feats, graph_feats,
                                                         W_if, b_if, W_iu, b_iu, indf, indu);
    k_flag<<<EDGES / 256, 256, 0, stream>>>(edge_feats, edge_refl, graph_feats,
                                            W_fl, b_fl, W_cls, b_cls, idx_revs, gnoise, flag0);

    // ---- BP loop (12 iterations, ping-pong) ----
    k_zero<<<(int)((BP_FLOATS + 255) / 256), 256, 0, stream>>>((float*)(ws + OFF_R), (int)BP_FLOATS);

    int cur = 0;
    for (int it = 0; it < NPG; ++it) {
        int nxt = 1 - cur;
        k_bp_edge<<<(2 * EDGES) / 256, 256, 0, stream>>>(
            fv, uvv,
            mmB[cur], umB[cur], msgB[cur], rmsgB[cur], umsgB[cur], rumsgB[cur],
            indf, indu, flag0, src, dst,
            msgB[nxt], rmsgB[nxt], umsgB[nxt], rumsgB[nxt]);
        k_aggregate<<<(NTOT * NACT) / 256, 256, 0, stream>>>(msgB[nxt], umsgB[nxt],
                                                             mmB[nxt], umB[nxt]);
        cur = nxt;
    }

    k_output<<<(NGRAPH * NACT) / 256, 256, 0, stream>>>(umB[cur], indu, (float*)d_out);
}
